// ParallelRetention_origin_25374666785438
// MI455X (gfx1250) — matrix operand order repaired and checked by reading, not compiled, not run on hardware
//
#include <hip/hip_runtime.h>

typedef __attribute__((ext_vector_type(16))) __bf16 v16bf;
typedef __attribute__((ext_vector_type(8)))  __bf16 v8bf;
typedef __attribute__((ext_vector_type(8)))  float  v8f;
typedef __attribute__((ext_vector_type(4)))  float  v4f;
typedef __attribute__((ext_vector_type(2)))  float  v2f;
typedef __attribute__((ext_vector_type(8)))  unsigned int v8u;

namespace {
constexpr int kH = 16;
constexpr int kS = 2048;
constexpr int kD = 64;
constexpr int kWaves = 4;   // waves (16-row tiles) per block
}

__device__ __forceinline__ unsigned short f2bf(float f) {
    unsigned int u = __float_as_uint(f);
    u += 0x7FFFu + ((u >> 16) & 1u);        // RNE (prep path only)
    return (unsigned short)(u >> 16);
}

__device__ __forceinline__ v16bf cat8(v8bf lo, v8bf hi) {
    return __builtin_shufflevector(lo, hi, 0,1,2,3,4,5,6,7,8,9,10,11,12,13,14,15);
}

// pack truncated bf16(flo), bf16(fhi) into one dword: (hi16(fhi)<<16)|hi16(flo)
__device__ __forceinline__ unsigned int pk_hi16(float fhi, float flo) {
    return __builtin_amdgcn_perm(__float_as_uint(fhi), __float_as_uint(flo), 0x07060302u);
}

// adjacent-pair tree reduce: v8f -> v2f (2 levels, 3x v_pk_add_f32, no gathers)
__device__ __forceinline__ v2f hsum8(v8f x) {
    v4f a = __builtin_shufflevector(x, x, 0, 1, 2, 3)
          + __builtin_shufflevector(x, x, 4, 5, 6, 7);
    return __builtin_shufflevector(a, a, 0, 1)
         + __builtin_shufflevector(a, a, 2, 3);
}

// Prep: q,k -> bf16 row-major [H][S][D]; v -> bf16 transposed [H][D][S]
__global__ void retention_prep(const float* __restrict__ q,
                               const float* __restrict__ k,
                               const float* __restrict__ v,
                               unsigned short* __restrict__ qb,
                               unsigned short* __restrict__ kb,
                               unsigned short* __restrict__ vtb) {
    int i = blockIdx.x * blockDim.x + threadIdx.x;
    int total = kH * kS * kD;
    if (i >= total) return;
    qb[i] = f2bf(q[i]);
    kb[i] = f2bf(k[i]);
    int s = i % kS;
    int d = (i / kS) % kD;
    int h = i / (kS * kD);
    vtb[i] = f2bf(v[(h * kS + s) * kD + d]);
}

__global__ void __launch_bounds__(kWaves * 32, 1)
retention_main(const unsigned short* __restrict__ qb,
               const unsigned short* __restrict__ kb,
               const unsigned short* __restrict__ vtb,
               const float* __restrict__ omask,
               float* __restrict__ out) {
    __shared__ float lds_u[kWaves][16 * 32];   // per-wave f32 qk-tile (2KB/wave)

    const int lane = threadIdx.x & 31;
    const int wave = threadIdx.x >> 5;
    const int l16  = lane & 15;
    const int g    = lane >> 4;          // lane half: 0 or 1

    const int h     = blockIdx.y;
    const int mbase = (blockIdx.x * kWaves + wave) * 16;

    const unsigned short* qh  = qb  + (size_t)h * kS * kD;
    const unsigned short* kh  = kb  + (size_t)h * kS * kD;
    const unsigned short* vth = vtb + (size_t)h * kD * kS;

    // ---- A operand (q tile, 16x64 bf16), loaded once.
    v16bf aq0, aq1;
    {
        const unsigned short* qrow = qh + (size_t)(mbase + l16) * kD;
        aq0 = cat8(*(const v8bf*)(qrow +      g * 8),
                   *(const v8bf*)(qrow + 16 + g * 8));
        aq1 = cat8(*(const v8bf*)(qrow + 32 + g * 8),
                   *(const v8bf*)(qrow + 48 + g * 8));
    }

    v8f accO[4] = {};                    // 16x64 f32 output tile (C-layout)
    v2f rs2 = {0.f, 0.f};                // omask row partials (row = l16)
    v2f ts2 = {0.f, 0.f};                // u row partials     (row = l16)

    float* myLds = lds_u[wave];
    const float* omrow = omask + ((size_t)h * kS + mbase + l16) * kS;

    #pragma unroll 1
    for (int j = 0; j < kS; j += 32) {
        if (j + 32 < kS) {  // next tile's row segment is exactly one 128B line
            __builtin_prefetch(omrow + (j + 32), 0, 0);
        }

        // ---- S = Q.K^T (two 16x16 n-subtiles)
        v8f sacc0 = {}, sacc1 = {};
        {
            const unsigned short* krow0 = kh + (size_t)(j + l16) * kD;
            const unsigned short* krow1 = krow0 + 16 * kD;
            v16bf bk00 = cat8(*(const v8bf*)(krow0 +      g * 8), *(const v8bf*)(krow0 + 16 + g * 8));
            v16bf bk01 = cat8(*(const v8bf*)(krow0 + 32 + g * 8), *(const v8bf*)(krow0 + 48 + g * 8));
            v16bf bk10 = cat8(*(const v8bf*)(krow1 +      g * 8), *(const v8bf*)(krow1 + 16 + g * 8));
            v16bf bk11 = cat8(*(const v8bf*)(krow1 + 32 + g * 8), *(const v8bf*)(krow1 + 48 + g * 8));
            sacc0 = __builtin_amdgcn_wmma_f32_16x16x32_bf16(false, aq0, false, bk00,
                                                            (short)0, sacc0, false, false);
            sacc1 = __builtin_amdgcn_wmma_f32_16x16x32_bf16(false, aq0, false, bk10,
                                                            (short)0, sacc1, false, false);
            sacc0 = __builtin_amdgcn_wmma_f32_16x16x32_bf16(false, aq1, false, bk01,
                                                            (short)0, sacc0, false, false);
            sacc1 = __builtin_amdgcn_wmma_f32_16x16x32_bf16(false, aq1, false, bk11,
                                                            (short)0, sacc1, false, false);
        }
        // spill raw f32; adjacent subtile stores pair into ds_store_2addr_b32
        #pragma unroll
        for (int i = 0; i < 8; ++i) {
            myLds[(i + 8 * g) * 32 +      l16] = sacc0[i];
            myLds[(i + 8 * g) * 32 + 16 + l16] = sacc1[i];
        }

        // ---- A-layout side: lane owns row l16, cols j+{g*8..+7, 16+g*8..+7}.
        const float* omp  = omrow + j;
        const float* urow = myLds + l16 * 32;    // same-wave DS ops: in-order
        v8f om0 = *(const v8f*)(omp  +      g * 8);
        v8f om1 = *(const v8f*)(omp  + 16 + g * 8);
        v8f qk0 = *(const v8f*)(urow +      g * 8);
        v8f qk1 = *(const v8f*)(urow + 16 + g * 8);

        v8f pr0 = om0 * qk0;                     // 4x v_pk_mul_f32, adjacent
        v8f pr1 = om1 * qk1;

        v8u apack;
        #pragma unroll
        for (int p = 0; p < 4; ++p) {
            apack[p]     = pk_hi16(pr0[2 * p + 1], pr0[2 * p]);  // truncating pack
            apack[4 + p] = pk_hi16(pr1[2 * p + 1], pr1[2 * p]);
        }
        v16bf au = __builtin_bit_cast(v16bf, apack);

        rs2 += hsum8(om0 + om1);                 // adjacent-pair pk_add tree
        ts2 += hsum8(pr0 + pr1);

        // ---- O += u @ v  (vT rows contiguous in s)
        #pragma unroll
        for (int dt = 0; dt < 4; ++dt) {
            const unsigned short* vrow = vth + (size_t)(dt * 16 + l16) * kS + j + g * 8;
            v16bf bv = cat8(*(const v8bf*)vrow, *(const v8bf*)(vrow + 16));
            accO[dt] = __builtin_amdgcn_wmma_f32_16x16x32_bf16(false, au, false, bv,
                                                               (short)0, accO[dt],
                                                               false, false);
        }
    }

    // ---- merge partials; lane l and l+16 hold complementary column sets
    float rsum = rs2.x + rs2.y;
    float tsum = ts2.x + ts2.y;
    rsum += __shfl_xor(rsum, 16, 32);
    tsum += __shfl_xor(tsum, 16, 32);

    // out = (u@v) / max(sqrt(r), |t|); 0 if denom==0 (nan_to_num path)
    #pragma unroll
    for (int i = 0; i < 8; ++i) {
        int row = i + 8 * g;                     // C-layout row for element i
        float r = __shfl(rsum, row, 32);
        float t = __shfl(tsum, row, 32);
        float denom = fmaxf(sqrtf(r), fabsf(t));
        float inv = denom > 0.f ? 1.f / denom : 0.f;
        float* orow = out + ((size_t)h * kS + mbase + row) * kD + l16;
        orow[0]  = accO[0][i] * inv;
        orow[16] = accO[1][i] * inv;
        orow[32] = accO[2][i] * inv;
        orow[48] = accO[3][i] * inv;
    }
}

extern "C" void kernel_launch(void* const* d_in, const int* in_sizes, int n_in,
                              void* d_out, int out_size, void* d_ws, size_t ws_size,
                              hipStream_t stream) {
    const float* q     = (const float*)d_in[0];
    const float* k     = (const float*)d_in[1];
    const float* v     = (const float*)d_in[2];
    const float* omask = (const float*)d_in[3];
    float* out = (float*)d_out;

    const int nQKV = kH * kS * kD;                       // 2,097,152 elems
    unsigned short* qb  = (unsigned short*)d_ws;         // 4 MB
    unsigned short* kb  = qb + nQKV;                     // 4 MB
    unsigned short* vtb = kb + nQKV;                     // 4 MB (transposed)

    retention_prep<<<(nQKV + 255) / 256, 256, 0, stream>>>(q, k, v, qb, kb, vtb);

    dim3 grid(kS / 16 / kWaves, kH);                     // (32, 16)
    retention_main<<<grid, kWaves * 32, 0, stream>>>(qb, kb, vtb, omask, out);
}
